// MultiHeadedAttention_11811160064156
// MI455X (gfx1250) — compile-verified, matrix operand order checked by reading, not run on hardware
//
#include <hip/hip_runtime.h>
#include <hip/hip_bf16.h>

typedef __bf16 bf16_t;
typedef __attribute__((ext_vector_type(16))) __bf16 v16bf;
typedef __attribute__((ext_vector_type(8)))  float  v8f;

static constexpr int B_ = 8, L_ = 1024, D_ = 1024, H_ = 16, DT_ = 64, DV_ = 64, DQK_ = 128;

// ---------------- fragment loaders (CDNA5 WMMA bf16 16x16x32 layouts) ----------------
// A-fragment: lane m(0..15)=row, K chunks {half*8..+7} and {16+half*8..+7}
// B-fragment (dual layout): lane n(0..15)=column, same K chunk split.
__device__ __forceinline__ v16bf frag_from_f32(const float* base, int ld) {
  const int l = threadIdx.x & 31;
  const int m = l & 15, half = (l >> 4) * 8;
  const float* p = base + (size_t)m * ld + half;
  v16bf r;
#pragma unroll
  for (int i = 0; i < 8; ++i) r[i] = (__bf16)p[i];
#pragma unroll
  for (int i = 0; i < 8; ++i) r[8 + i] = (__bf16)p[16 + i];
  return r;
}

__device__ __forceinline__ v16bf frag_from_bf16(const bf16_t* base, int ld) {
  const int l = threadIdx.x & 31;
  const int m = l & 15, half = (l >> 4) * 8;
  const bf16_t* p = base + (size_t)m * ld + half;
  v16bf r;
#pragma unroll
  for (int i = 0; i < 8; ++i) r[i] = p[i];
#pragma unroll
  for (int i = 0; i < 8; ++i) r[8 + i] = p[16 + i];
  return r;
}

// ---------------- prep kernels ----------------
__global__ void __launch_bounds__(256) k_f32_to_bf16(const float* __restrict__ src,
                                                     bf16_t* __restrict__ dst, int n) {
  int i = blockIdx.x * blockDim.x + threadIdx.x;
  const int stride = gridDim.x * blockDim.x;
  for (; i < n; i += stride) dst[i] = (__bf16)src[i];
}

// pack mask (B,1,L,L) int32 -> 1 bit per entry (bit=1 => masked out): 32x traffic cut
__global__ void __launch_bounds__(256) k_pack_mask(const int* __restrict__ mask,
                                                   unsigned* __restrict__ bits, int total) {
  int i = blockIdx.x * blockDim.x + threadIdx.x;
  const int stride = gridDim.x * blockDim.x;
  for (; i < total; i += stride) {
    const int kw = i & 31;            // which 32-key word in the row
    const int rowg = i >> 5;          // b*L + row
    const int* mp = mask + (size_t)rowg * L_ + kw * 32;
    unsigned v = 0;
#pragma unroll
    for (int j = 0; j < 32; ++j) v |= (mp[j] == 1 ? 1u : 0u) << j;
    bits[i] = v;
  }
}

// ---------------- projection GEMM: C = X @ W^T + bias ----------------
// Block = 8 waves sharing one 16-row A stripe: the 16x32 f32 A tile is converted to
// bf16 and staged in LDS once per k-step (8x fewer loads + conversions than per-wave).
// mode 0/1: write head-split into (B,H,L,128) at features [0,64)
// mode 2  : write transposed into (B,H,64,L)
__global__ void __launch_bounds__(256) k_proj(const float* __restrict__ X,
                                              const bf16_t* __restrict__ Wb,
                                              const float* __restrict__ bias,
                                              bf16_t* __restrict__ dst, int mode) {
  __shared__ bf16_t As[16][32];                      // 1 KB staged A tile
  const int tid = threadIdx.x;
  const int wv = tid >> 5;
  const int blk = blockIdx.x;                        // 4096 blocks
  const int tm = blk >> 3;                           // 512 row-tiles
  const int tn = (blk & 7) * 8 + wv;                 // 64 col-tiles
  const float*  Abase = X  + (size_t)tm * 16 * D_;
  const bf16_t* Bbase = Wb + (size_t)tn * 16 * D_;
  v8f acc = {};
  for (int k0 = 0; k0 < D_; k0 += 32) {
    // cooperative stage: 512 f32 -> bf16, 2 elements per thread
    {
      const int i0 = tid, i1 = tid + 256;
      As[i0 >> 5][i0 & 31] = (__bf16)Abase[(size_t)(i0 >> 5) * D_ + k0 + (i0 & 31)];
      As[i1 >> 5][i1 & 31] = (__bf16)Abase[(size_t)(i1 >> 5) * D_ + k0 + (i1 & 31)];
    }
    if (k0 + 32 < D_) {
      __builtin_prefetch(Abase + ((tid >> 5) & 15) * D_ + k0 + 32, 0, 3);
      __builtin_prefetch(Bbase + ((tid >> 5) & 15) * D_ + k0 + 32, 0, 3);
    }
    __syncthreads();
    v16bf a = frag_from_bf16(&As[0][0], 32);
    v16bf b = frag_from_bf16(Bbase + k0, D_);
    acc = __builtin_amdgcn_wmma_f32_16x16x32_bf16(false, a, false, b, (short)0, acc, false, false);
    __syncthreads();
  }
  const int l = tid & 31;
  const int col = tn * 16 + (l & 15);
  const int rbase = (l >> 4) * 8;                    // C layout: lanes 16..31 hold rows 8..15
  const float bv = bias[col];
  const int h = col >> 6, dv = col & 63;
#pragma unroll
  for (int r = 0; r < 8; ++r) {
    const int m = tm * 16 + rbase + r;
    const int bi = m >> 10, li = m & (L_ - 1);
    const float v = acc[r] + bv;
    if (mode == 2)
      dst[((size_t)(bi * H_ + h) * DV_ + dv) * L_ + li] = (__bf16)v;
    else
      dst[((size_t)(bi * H_ + h) * L_ + li) * DQK_ + dv] = (__bf16)v;
  }
}

// temporal projection t = T @ WT^T + bT, broadcast into qc/kc features [64,128) for all heads
__global__ void __launch_bounds__(256) k_proj_t(const float* __restrict__ T,
                                                const bf16_t* __restrict__ WTb,
                                                const float* __restrict__ bT,
                                                bf16_t* __restrict__ qc,
                                                bf16_t* __restrict__ kc) {
  const int wid = (blockIdx.x * blockDim.x + threadIdx.x) >> 5;
  const int tiles_n = DT_ / 16;                      // 4
  const int total = (B_ * L_ / 16) * tiles_n;        // 2048 waves
  if (wid >= total) return;
  const int tm = wid / tiles_n, tn = wid % tiles_n;
  const float*  Abase = T   + (size_t)tm * 16 * DT_;
  const bf16_t* Bbase = WTb + (size_t)tn * 16 * DT_;
  v8f acc = {};
  for (int k0 = 0; k0 < DT_; k0 += 32) {
    v16bf a = frag_from_f32(Abase + k0, DT_);
    v16bf b = frag_from_bf16(Bbase + k0, DT_);
    acc = __builtin_amdgcn_wmma_f32_16x16x32_bf16(false, a, false, b, (short)0, acc, false, false);
  }
  const int l = threadIdx.x & 31;
  const int col = tn * 16 + (l & 15);                // 0..63
  const int rbase = (l >> 4) * 8;
  const float bv = bT[col];
#pragma unroll
  for (int r = 0; r < 8; ++r) {
    const int m = tm * 16 + rbase + r;
    const int bi = m >> 10, li = m & (L_ - 1);
    const bf16_t v = (__bf16)(acc[r] + bv);
#pragma unroll
    for (int h = 0; h < H_; ++h) {
      const size_t off = ((size_t)(bi * H_ + h) * L_ + li) * DQK_ + DV_ + col;
      qc[off] = v;
      kc[off] = v;
    }
  }
}

// ---------------- flash-attention ----------------
// One block (8 waves) = 8 consecutive 16-query tiles of one (b,h). Each 32-key step
// stages K (32x128) and V (64x32) tiles into LDS cooperatively -> 8x less K/V traffic.
__global__ void __launch_bounds__(256) k_attn(const bf16_t* __restrict__ qc,
                                              const bf16_t* __restrict__ kc,
                                              const bf16_t* __restrict__ vT,
                                              const unsigned* __restrict__ mbits,
                                              bf16_t* __restrict__ xb) {
  __shared__ bf16_t Kt[32][DQK_];                    // 8 KB: 32 keys x 128 features
  __shared__ bf16_t Vt[DV_][32];                     // 4 KB: 64 dv x 32 keys
  __shared__ bf16_t Pt[8][16][32];                   // 8 KB: per-wave P patch
  const int tid = threadIdx.x;
  const int wv = tid >> 5;
  const int l = tid & 31;
  const int c = l & 15;
  const int rbase = (l >> 4) * 8;

  const int blk = blockIdx.x;                        // B*H*(L/128) = 1024
  const int qblocks = L_ / 128;                      // 8
  const int qt = (blk % qblocks) * 8 + wv;           // this wave's 16-query tile
  const int h  = (blk / qblocks) % H_;
  const int bi = blk / (qblocks * H_);

  const bf16_t* qh = qc + (size_t)(bi * H_ + h) * L_ * DQK_;
  const bf16_t* kh = kc + (size_t)(bi * H_ + h) * L_ * DQK_;
  const bf16_t* vh = vT + (size_t)(bi * H_ + h) * DV_ * L_;
  const unsigned* mb = mbits + (size_t)bi * L_ * (L_ / 32);

  v16bf qf[4];
#pragma unroll
  for (int kk = 0; kk < 4; ++kk)
    qf[kk] = frag_from_bf16(qh + (size_t)qt * 16 * DQK_ + kk * 32, DQK_);

  v8f O0 = {}, O1 = {}, O2 = {}, O3 = {};
  float mcur[8], lsum[8];
#pragma unroll
  for (int r = 0; r < 8; ++r) { mcur[r] = -3.0e38f; lsum[r] = 0.f; }

  const float sc = 0.08838834764831845f;             // 1/sqrt(128)

  for (int kb = 0; kb < L_; kb += 32) {
    // ---- cooperative staging of K and V tiles (16B vector copies) ----
    {
      const uint4* ksrc = (const uint4*)(kh + (size_t)kb * DQK_);   // 32*128*2B = 512 uint4
      uint4* kdst = (uint4*)&Kt[0][0];
      kdst[tid]       = ksrc[tid];
      kdst[tid + 256] = ksrc[tid + 256];
      const int dv = tid >> 2, part = tid & 3;                      // 64 rows x 4 uint4
      const uint4* vsrc = (const uint4*)(vh + (size_t)dv * L_ + kb);
      ((uint4*)&Vt[dv][0])[part] = vsrc[part];
    }
    if (kb + 32 < L_) {
      __builtin_prefetch(kh + (size_t)(kb + 32) * DQK_ + tid * 8, 0, 3);
      __builtin_prefetch(vh + (size_t)(tid >> 2) * L_ + kb + 32, 0, 3);
    }
    __syncthreads();

    // ---- S tiles: 16 queries x 32 keys, K from LDS ----
    v8f S0 = {}, S1 = {};
#pragma unroll
    for (int kk = 0; kk < 4; ++kk) {
      v16bf b0 = frag_from_bf16(&Kt[0][kk * 32], DQK_);
      v16bf b1 = frag_from_bf16(&Kt[16][kk * 32], DQK_);
      S0 = __builtin_amdgcn_wmma_f32_16x16x32_bf16(false, qf[kk], false, b0, (short)0, S0, false, false);
      S1 = __builtin_amdgcn_wmma_f32_16x16x32_bf16(false, qf[kk], false, b1, (short)0, S1, false, false);
    }

    // ---- scale + bit-packed mask + online softmax per row ----
    float alpha[8];
#pragma unroll
    for (int r = 0; r < 8; ++r) {
      const int row = qt * 16 + rbase + r;
      const unsigned w = mb[(size_t)row * (L_ / 32) + (kb >> 5)];
      float s0 = ((w >> c) & 1u)        ? -1.0e9f : S0[r] * sc;
      float s1 = ((w >> (16 + c)) & 1u) ? -1.0e9f : S1[r] * sc;
      float mx = fmaxf(s0, s1);
      mx = fmaxf(mx, __shfl_xor(mx, 1, 32));
      mx = fmaxf(mx, __shfl_xor(mx, 2, 32));
      mx = fmaxf(mx, __shfl_xor(mx, 4, 32));
      mx = fmaxf(mx, __shfl_xor(mx, 8, 32));         // row max over the 16-lane half
      const float mn = fmaxf(mcur[r], mx);
      const float a  = __expf(mcur[r] - mn);
      const float p0 = __expf(s0 - mn);
      const float p1 = __expf(s1 - mn);
      float s = p0 + p1;
      s += __shfl_xor(s, 1, 32);
      s += __shfl_xor(s, 2, 32);
      s += __shfl_xor(s, 4, 32);
      s += __shfl_xor(s, 8, 32);
      lsum[r] = lsum[r] * a + s;
      mcur[r] = mn;
      alpha[r] = a;
      // C-layout -> per-wave LDS patch (row-major 16x32) for A-layout reload
      Pt[wv][rbase + r][c]      = (__bf16)p0;
      Pt[wv][rbase + r][16 + c] = (__bf16)p1;
    }
#pragma unroll
    for (int r = 0; r < 8; ++r) {
      O0[r] *= alpha[r]; O1[r] *= alpha[r]; O2[r] *= alpha[r]; O3[r] *= alpha[r];
    }
    asm volatile("s_wait_dscnt 0" ::: "memory");     // cross-lane LDS visibility within wave

    // ---- PV: P (16x32) @ V (32x64), V from LDS ----
    v16bf pf = frag_from_bf16(&Pt[wv][0][0], 32);
    v16bf v0 = frag_from_bf16(&Vt[0][0],  32);
    v16bf v1 = frag_from_bf16(&Vt[16][0], 32);
    v16bf v2 = frag_from_bf16(&Vt[32][0], 32);
    v16bf v3 = frag_from_bf16(&Vt[48][0], 32);
    O0 = __builtin_amdgcn_wmma_f32_16x16x32_bf16(false, pf, false, v0, (short)0, O0, false, false);
    O1 = __builtin_amdgcn_wmma_f32_16x16x32_bf16(false, pf, false, v1, (short)0, O1, false, false);
    O2 = __builtin_amdgcn_wmma_f32_16x16x32_bf16(false, pf, false, v2, (short)0, O2, false, false);
    O3 = __builtin_amdgcn_wmma_f32_16x16x32_bf16(false, pf, false, v3, (short)0, O3, false, false);
    __syncthreads();                                 // Kt/Vt consumed before next staging
  }

  // ---- finalize: divide by row sums, write x (B, L, D) bf16 ----
#pragma unroll
  for (int r = 0; r < 8; ++r) {
    const float inv = 1.0f / lsum[r];
    O0[r] *= inv; O1[r] *= inv; O2[r] *= inv; O3[r] *= inv;
  }
#pragma unroll
  for (int r = 0; r < 8; ++r) {
    const int li = qt * 16 + rbase + r;
    const size_t base = ((size_t)bi * L_ + li) * D_ + (size_t)h * DV_;
    xb[base + 0 * 16 + c] = (__bf16)O0[r];
    xb[base + 1 * 16 + c] = (__bf16)O1[r];
    xb[base + 2 * 16 + c] = (__bf16)O2[r];
    xb[base + 3 * 16 + c] = (__bf16)O3[r];
  }
}

// ---------------- output projection: out = x @ WM^T + bM (fp32 out) ----------------
// Same block-shared A staging as k_proj (A already bf16 -> plain 16B copies).
__global__ void __launch_bounds__(256) k_outproj(const bf16_t* __restrict__ xb,
                                                 const bf16_t* __restrict__ WMb,
                                                 const float* __restrict__ bM,
                                                 float* __restrict__ out) {
  __shared__ bf16_t As[16][32];
  const int tid = threadIdx.x;
  const int wv = tid >> 5;
  const int blk = blockIdx.x;
  const int tm = blk >> 3;
  const int tn = (blk & 7) * 8 + wv;
  const bf16_t* Abase = xb  + (size_t)tm * 16 * D_;
  const bf16_t* Bbase = WMb + (size_t)tn * 16 * D_;
  v8f acc = {};
  for (int k0 = 0; k0 < D_; k0 += 32) {
    if (tid < 64) {                                  // 16x32 bf16 = 64 uint4
      const int row = tid >> 2, part = tid & 3;
      const uint4* src = (const uint4*)(Abase + (size_t)row * D_ + k0);
      ((uint4*)&As[row][0])[part] = src[part];
    }
    if (k0 + 32 < D_)
      __builtin_prefetch(Bbase + ((tid >> 5) & 15) * D_ + k0 + 32, 0, 3);
    __syncthreads();
    v16bf a = frag_from_bf16(&As[0][0], 32);
    v16bf b = frag_from_bf16(Bbase + k0, D_);
    acc = __builtin_amdgcn_wmma_f32_16x16x32_bf16(false, a, false, b, (short)0, acc, false, false);
    __syncthreads();
  }
  const int l = tid & 31;
  const int col = tn * 16 + (l & 15);
  const int rbase = (l >> 4) * 8;
  const float bv = bM[col];
#pragma unroll
  for (int r = 0; r < 8; ++r) {
    const int m = tm * 16 + rbase + r;
    out[(size_t)m * D_ + col] = acc[r] + bv;
  }
}

// ---------------- host launch ----------------
extern "C" void kernel_launch(void* const* d_in, const int* in_sizes, int n_in,
                              void* d_out, int out_size, void* d_ws, size_t ws_size,
                              hipStream_t stream) {
  const float* Q  = (const float*)d_in[0];
  const float* K  = (const float*)d_in[1];
  const float* V  = (const float*)d_in[2];
  const float* T  = (const float*)d_in[3];
  const int*  mask = (const int*)d_in[4];
  const float* WQ = (const float*)d_in[5];
  const float* bQ = (const float*)d_in[6];
  const float* WK = (const float*)d_in[7];
  const float* bK = (const float*)d_in[8];
  const float* WV = (const float*)d_in[9];
  const float* bV = (const float*)d_in[10];
  const float* WM = (const float*)d_in[11];
  const float* bM = (const float*)d_in[12];
  const float* WT = (const float*)d_in[13];
  const float* bT = (const float*)d_in[14];
  float* out = (float*)d_out;

  // workspace layout (bf16 regions first, then packed mask words) — ~105 MB total
  bf16_t* p = (bf16_t*)d_ws;
  bf16_t* WQb = p; p += (size_t)D_ * D_;
  bf16_t* WKb = p; p += (size_t)D_ * D_;
  bf16_t* WVb = p; p += (size_t)D_ * D_;
  bf16_t* WMb = p; p += (size_t)D_ * D_;
  bf16_t* WTb = p; p += (size_t)DT_ * DT_;
  bf16_t* qc  = p; p += (size_t)B_ * H_ * L_ * DQK_;   // (B,H,L,128)
  bf16_t* kc  = p; p += (size_t)B_ * H_ * L_ * DQK_;   // (B,H,L,128)
  bf16_t* vT  = p; p += (size_t)B_ * H_ * DV_ * L_;    // (B,H,64,L)
  bf16_t* xb  = p; p += (size_t)B_ * L_ * D_;          // (B,L,D)
  unsigned* bits = (unsigned*)p;                        // B*L*32 words

  // weight conversion + mask bit-pack
  k_f32_to_bf16<<<1024, 256, 0, stream>>>(WQ, WQb, D_ * D_);
  k_f32_to_bf16<<<1024, 256, 0, stream>>>(WK, WKb, D_ * D_);
  k_f32_to_bf16<<<1024, 256, 0, stream>>>(WV, WVb, D_ * D_);
  k_f32_to_bf16<<<1024, 256, 0, stream>>>(WM, WMb, D_ * D_);
  k_f32_to_bf16<<<16,   256, 0, stream>>>(WT, WTb, DT_ * DT_);
  k_pack_mask<<<1024, 256, 0, stream>>>(mask, bits, B_ * L_ * 32);

  // projections (8 waves/block; block shares one A stripe via LDS)
  k_proj<<<4096, 256, 0, stream>>>(Q, WQb, bQ, qc, 0);
  k_proj<<<4096, 256, 0, stream>>>(K, WKb, bK, kc, 1);
  k_proj<<<4096, 256, 0, stream>>>(V, WVb, bV, vT, 2);
  k_proj_t<<<256, 256, 0, stream>>>(T, WTb, bT, qc, kc);

  // flash attention (block shares K/V tiles via LDS) + output projection
  k_attn<<<1024, 256, 0, stream>>>(qc, kc, vT, bits, xb);
  k_outproj<<<4096, 256, 0, stream>>>(xb, WMb, bM, out);
}
